// LHSBv2_40381282517306
// MI455X (gfx1250) — matrix operand order repaired and checked by reading, not compiled, not run on hardware
//
#include <hip/hip_runtime.h>
#include <math.h>

// ---------------------------------------------------------------- types
typedef __bf16 bf16;
typedef __attribute__((ext_vector_type(16))) __bf16       v16bf;
typedef __attribute__((ext_vector_type(8)))  float        v8f;
typedef __attribute__((ext_vector_type(4)))  unsigned int v4u;

#define NWIN 4096   // total windows (B*gh*gw = 4*32*32)
#define CDIM 96
#define PTOK 64     // tokens per window (8*8)
#define HW   256

union FragU { v16bf f; v4u u[2]; };
union Pack8 { v4u u; bf16 h[8]; };

// A fragment: 16x32 (MxK) bf16, source row-major [16+][ld]
// lane L<16: row L, K = {0..7, 16..23}; lane L+16: row L-16, K = {8..15, 24..31}
static __device__ __forceinline__ v16bf load_frag_a(const bf16* base, int ld) {
    const int lane = threadIdx.x & 31;
    const int hf = lane >> 4, r = lane & 15;
    const bf16* p = base + r * ld + hf * 8;
    FragU t;
    t.u[0] = *(const v4u*)(p);
    t.u[1] = *(const v4u*)(p + 16);
    return t.f;
}

// B fragment for D = A x Y^T: Y row-major [16+ rows(N)][ld(K)]
// lane L holds Y row (L&15); lanes<16 carry K {0..7, 16..23}, lanes>=16 carry K {8..15, 24..31}
static __device__ __forceinline__ v16bf load_frag_b(const bf16* base, int ld) {
    const int lane = threadIdx.x & 31;
    const int hf = lane >> 4, r = lane & 15;
    const bf16* p = base + r * ld + hf * 8;
    FragU t;
    t.u[0] = *(const v4u*)(p);
    t.u[1] = *(const v4u*)(p + 16);
    return t.f;
}

#define WMMA_BF16(a, b, c) \
    __builtin_amdgcn_wmma_f32_16x16x32_bf16(false, (a), false, (b), (short)0, (c), false, false)

// Bitwise blend: cannot be lowered to control flow (no select in IR).
static __device__ __forceinline__ int isel(bool c, int a, int b) {
    const int m = -(int)c;            // all-ones if c
    return (a & m) | (b & ~m);        // folds to v_bfi/v_cndmask
}

// Branchless sorted top-3 insert (invariant t0>=t1>=t2):
// candidate competes with slot 2 (v_max/v_min), then compare-swaps upward.
static __device__ __forceinline__ void top3_update(float v, int ix,
                                                   float& t0, float& t1, float& t2,
                                                   int& j0, int& j1, int& j2) {
    const bool g2 = v > t2;                      // slot-2 compete
    t2 = fmaxf(t2, v);
    j2 = isel(g2, ix, j2);
    const bool g1 = t2 > t1;                     // bubble 2 -> 1
    const float h1 = fmaxf(t1, t2), l1 = fminf(t1, t2);
    const int  jh1 = isel(g1, j2, j1), jl1 = isel(g1, j1, j2);
    t1 = h1; t2 = l1; j1 = jh1; j2 = jl1;
    const bool g0 = t1 > t0;                     // bubble 1 -> 0
    const float h0 = fmaxf(t0, t1), l0 = fminf(t0, t1);
    const int  jh0 = isel(g0, j1, j0), jl0 = isel(g0, j0, j1);
    t0 = h0; t1 = l0; j0 = jh0; j1 = jl0;
}

// ------------------------------------------------ kernel 1: 8x8 max-pool -> bf16 dsx [NWIN][96]
__global__ void pool_kernel(const float* __restrict__ x, bf16* __restrict__ dsxh) {
    int tid = blockIdx.x * 256 + threadIdx.x;      // c*4096 + n
    int c = tid >> 12, n = tid & 4095;
    int b = n >> 10, gy = (n >> 5) & 31, gx = n & 31;
    const float* p = x + (((size_t)(b * CDIM + c) * HW + gy * 8) * HW + gx * 8);
    float m = -1e30f;
#pragma unroll
    for (int dy = 0; dy < 8; dy++) {
        const float4* pr = (const float4*)(p + dy * HW);
        float4 u = pr[0], v = pr[1];
        m = fmaxf(m, fmaxf(fmaxf(u.x, u.y), fmaxf(u.z, u.w)));
        m = fmaxf(m, fmaxf(fmaxf(v.x, v.y), fmaxf(v.z, v.w)));
    }
    dsxh[n * CDIM + c] = (bf16)m;
}

// ------------------------------------------------ kernel 2: sim = dsx dsx^T, global top-3 + softmax weights
// grid 256 blocks (16 rows each), block 128 (4 waves); wave w covers col-tiles w, w+4, ...
// Double-buffered B fragments + L2 prefetch 8 tiles ahead; branchless per-lane top-3.
__global__ __launch_bounds__(128) void sim_topk_kernel(const bf16* __restrict__ dsxh,
                                                       int* __restrict__ topk_idx,
                                                       float* __restrict__ topk_wgt) {
    __shared__ float s_val[16][64 * 3];
    __shared__ int   s_idx[16][64 * 3];

    const int tid = threadIdx.x, wave = tid >> 5, lane = tid & 31;
    const int hf = lane >> 4, lr = lane & 15;
    const int m0 = blockIdx.x * 16;

    const v16bf a0 = load_frag_a(dsxh + (size_t)m0 * CDIM + 0,  CDIM);
    const v16bf a1 = load_frag_a(dsxh + (size_t)m0 * CDIM + 32, CDIM);
    const v16bf a2 = load_frag_a(dsxh + (size_t)m0 * CDIM + 64, CDIM);

    float t0[8], t1[8], t2[8];
    int   j0[8], j1[8], j2[8];
#pragma unroll
    for (int j = 0; j < 8; j++) { t0[j] = t1[j] = t2[j] = -1e30f; j0[j] = j1[j] = j2[j] = 0; }

    const int l24 = lane - (lane >= 24 ? 24 : 0);   // lane % 24 without a branch
    // running prefetch pointer: tile (wave + 32), one 128B line per lane (24 lines/tile)
    const char* pfp = (const char*)dsxh + (size_t)((wave + 32) * 16) * CDIM * 2 + l24 * 128;

    v16bf b0, b1, b2;
    {
        const bf16* p = dsxh + (size_t)(wave * 16) * CDIM;
        b0 = load_frag_b(p + 0, CDIM);
        b1 = load_frag_b(p + 32, CDIM);
        b2 = load_frag_b(p + 64, CDIM);
    }
    for (int tt = 0; tt < 64; tt++) {
        const int cur = wave + tt * 4;
        if (tt < 56)                     // prefetch tile 8 iterations ahead into L2
            __builtin_prefetch(pfp, 0, 2);
        pfp += 4 * 16 * CDIM * 2;
        v16bf nb0 = b0, nb1 = b1, nb2 = b2;
        if (tt < 63) {
            const bf16* p = dsxh + (size_t)((cur + 4) * 16) * CDIM;
            nb0 = load_frag_b(p + 0, CDIM);
            nb1 = load_frag_b(p + 32, CDIM);
            nb2 = load_frag_b(p + 64, CDIM);
        }
        v8f c = {};
        c = WMMA_BF16(a0, b0, c);
        c = WMMA_BF16(a1, b1, c);
        c = WMMA_BF16(a2, b2, c);
        const int col = cur * 16 + lr;
#pragma unroll
        for (int j = 0; j < 8; j++)
            top3_update(c[j], col, t0[j], t1[j], t2[j], j0[j], j1[j], j2[j]);
        b0 = nb0; b1 = nb1; b2 = nb2;
    }
#pragma unroll
    for (int j = 0; j < 8; j++) {
        const int r = j + hf * 8;
        const int base = (wave * 16 + lr) * 3;
        s_val[r][base + 0] = t0[j]; s_idx[r][base + 0] = j0[j];
        s_val[r][base + 1] = t1[j]; s_idx[r][base + 1] = j1[j];
        s_val[r][base + 2] = t2[j]; s_idx[r][base + 2] = j2[j];
    }
    __syncthreads();

    if (tid < 16) {
        float b0v = -1e30f, b1v = -1e30f, b2v = -1e30f;
        int i0 = 0, i1 = 0, i2 = 0;
        for (int s = 0; s < 192; s++)
            top3_update(s_val[tid][s], s_idx[tid][s], b0v, b1v, b2v, i0, i1, i2);
        const float e0 = 1.0f, e1 = __expf(b1v - b0v), e2 = __expf(b2v - b0v);
        const float inv = 1.0f / (e0 + e1 + e2);
        const int row = m0 + tid;
        topk_idx[row * 3 + 0] = i0; topk_wgt[row * 3 + 0] = e0 * inv;
        topk_idx[row * 3 + 1] = i1; topk_wgt[row * 3 + 1] = e1 * inv;
        topk_idx[row * 3 + 2] = i2; topk_wgt[row * 3 + 2] = e2 * inv;
    }
}

// ------------------------------------------------ kernel 3: window partition x -> bf16 xwh [NWIN][64][96]
// float4-coalesced reads; 4 consecutive w stay inside one window row.
__global__ void partition_kernel(const float* __restrict__ x, bf16* __restrict__ xwh) {
    size_t t4 = ((size_t)blockIdx.x * 256 + threadIdx.x) * 4;   // flat over [b][c][h][w]
    float4 v = *(const float4*)(x + t4);
    int w = (int)(t4 & 255), h = (int)((t4 >> 8) & 255);
    int bc = (int)(t4 >> 16), c = bc % CDIM, b = bc / CDIM;
    int gy = h >> 3, py = h & 7, gx = w >> 3, px = w & 7;
    int n = (b * 32 + gy) * 32 + gx;
    size_t base = ((size_t)n * PTOK + py * 8 + px) * CDIM + c;
    xwh[base + 0 * CDIM] = (bf16)v.x;
    xwh[base + 1 * CDIM] = (bf16)v.y;
    xwh[base + 2 * CDIM] = (bf16)v.z;
    xwh[base + 3 * CDIM] = (bf16)v.w;
}

// ------------------------------------------------ kernel 4: weights -> transposed bf16 Wh [4][96][96]
__global__ void wconv_kernel(const float* __restrict__ Wq, const float* __restrict__ Wk,
                             const float* __restrict__ Wv, const float* __restrict__ Wp,
                             bf16* __restrict__ Wh) {
    int tid = blockIdx.x * 256 + threadIdx.x;              // over 4*9216
    int sel = tid / 9216, r = tid % 9216;
    int k = r / CDIM, ncol = r % CDIM;
    const float* W = sel == 0 ? Wq : sel == 1 ? Wk : sel == 2 ? Wv : Wp;
    Wh[sel * 9216 + ncol * CDIM + k] = (bf16)W[k * CDIM + ncol];   // store W^T
}

// ------------------------------------------------ kernel 5: fused per-window attention
__global__ __launch_bounds__(128) void attn_kernel(
    const bf16* __restrict__ xwh, const bf16* __restrict__ Wh,
    const int* __restrict__ topk_idx, const float* __restrict__ topk_wgt,
    const float* __restrict__ bq, const float* __restrict__ bk,
    const float* __restrict__ bv, const float* __restrict__ bp,
    const float* __restrict__ lepe_w, const float* __restrict__ lepe_b,
    float* __restrict__ out) {
    __shared__ __align__(32) bf16  s_xw[PTOK * CDIM];
    __shared__ __align__(32) bf16  s_ctx[PTOK * CDIM];
    __shared__ __align__(32) bf16  s_q[PTOK * CDIM];
    __shared__ __align__(32) bf16  s_k[PTOK * CDIM];
    __shared__ __align__(32) bf16  s_vT[CDIM * PTOK];
    __shared__ __align__(32) float s_attn[PTOK * PTOK];
    __shared__ __align__(32) bf16  s_attnh[PTOK * PTOK];
    __shared__ __align__(32) bf16  s_o[PTOK * CDIM];
    __shared__ float s_lw[CDIM * 9];
    __shared__ float s_red[128];

    const int n = blockIdx.x;
    const int tid = threadIdx.x, wave = tid >> 5, lane = tid & 31;
    const int hf = lane >> 4, lr = lane & 15;

    // pull the weight block toward this WGP (all four W^T matrices, 72KB; L2-resident)
    for (int off = tid * 128; off < 4 * 9216 * (int)sizeof(bf16); off += 128 * 128)
        __builtin_prefetch((const char*)Wh + off, 0, 2);

    const int i0 = topk_idx[n * 3 + 0], i1 = topk_idx[n * 3 + 1], i2 = topk_idx[n * 3 + 2];
    const float w0 = topk_wgt[n * 3 + 0], w1 = topk_wgt[n * 3 + 1], w2 = topk_wgt[n * 3 + 2];
    const v4u* xn4 = (const v4u*)(xwh + (size_t)n  * (PTOK * CDIM));
    const v4u* c04 = (const v4u*)(xwh + (size_t)i0 * (PTOK * CDIM));
    const v4u* c14 = (const v4u*)(xwh + (size_t)i1 * (PTOK * CDIM));
    const v4u* c24 = (const v4u*)(xwh + (size_t)i2 * (PTOK * CDIM));
    v4u* sxw4 = (v4u*)s_xw;
    v4u* sctx4 = (v4u*)s_ctx;

    for (int i = tid; i < CDIM * 9; i += 128) s_lw[i] = lepe_w[i];
#pragma unroll
    for (int i = tid; i < (PTOK * CDIM) / 8; i += 128) {   // 6 iterations of b128 traffic
        Pack8 a; a.u = xn4[i];
        sxw4[i] = a.u;
        Pack8 p0, p1, p2, r;
        p0.u = c04[i]; p1.u = c14[i]; p2.u = c24[i];
#pragma unroll
        for (int e = 0; e < 8; e++)
            r.h[e] = (bf16)(w0 * (float)p0.h[e] + w1 * (float)p1.h[e] + w2 * (float)p2.h[e]);
        sctx4[i] = r.u;
    }
    __syncthreads();

    // ---- q = xw@Wq^T' + bq ; k = ctx@Wk ; v = ctx@Wv (v stored transposed)
    for (int job = wave; job < 72; job += 4) {
        const int mat = job / 24, t = job % 24, mt = t / 6, nt = t % 6;
        const bf16* A  = (mat == 0 ? s_xw : s_ctx) + mt * 16 * CDIM;
        const bf16* Bm = Wh + mat * 9216 + nt * 16 * CDIM;
        const float* bias = (mat == 0 ? bq : mat == 1 ? bk : bv);
        v8f c = {};
#pragma unroll
        for (int k0 = 0; k0 < CDIM; k0 += 32)
            c = WMMA_BF16(load_frag_a(A + k0, CDIM), load_frag_b(Bm + k0, CDIM), c);
        const int col = nt * 16 + lr;
        const float bb = bias[col];
#pragma unroll
        for (int j = 0; j < 8; j++) {
            const int row = mt * 16 + j + hf * 8;
            const float v = c[j] + bb;
            if (mat == 0)      s_q[row * CDIM + col]  = (bf16)v;
            else if (mat == 1) s_k[row * CDIM + col]  = (bf16)v;
            else               s_vT[col * PTOK + row] = (bf16)v;
        }
    }
    __syncthreads();

    // ---- scores = q k^T * scale
    const float scale = 0.10206207f;   // 96^-0.5
    for (int job = wave; job < 16; job += 4) {
        const int mt = job / 4, nt = job % 4;
        v8f c = {};
#pragma unroll
        for (int k0 = 0; k0 < CDIM; k0 += 32)
            c = WMMA_BF16(load_frag_a(s_q + mt * 16 * CDIM + k0, CDIM),
                          load_frag_b(s_k + nt * 16 * CDIM + k0, CDIM), c);
        const int col = nt * 16 + lr;
#pragma unroll
        for (int j = 0; j < 8; j++)
            s_attn[(mt * 16 + j + hf * 8) * PTOK + col] = c[j] * scale;
    }
    __syncthreads();

    // ---- row softmax -> bf16 probs (2 threads per row, pair-reduce through LDS)
    {
        const int row = tid >> 1, off = (tid & 1) * 32;
        const float* rowp = s_attn + row * PTOK + off;
        float ev[32];
        float mx = -1e30f;
#pragma unroll
        for (int q = 0; q < 32; q++) mx = fmaxf(mx, rowp[q]);
        s_red[tid] = mx;
        __syncthreads();
        mx = fmaxf(s_red[tid], s_red[tid ^ 1]);
        __syncthreads();
        float s = 0.f;
#pragma unroll
        for (int q = 0; q < 32; q++) { ev[q] = __expf(rowp[q] - mx); s += ev[q]; }
        s_red[tid] = s;
        __syncthreads();
        const float inv = 1.f / (s_red[tid] + s_red[tid ^ 1]);
#pragma unroll
        for (int q = 0; q < 32; q++) s_attnh[row * PTOK + off + q] = (bf16)(ev[q] * inv);
    }
    __syncthreads();

    // ---- o = attn @ v + lepe
    for (int job = wave; job < 24; job += 4) {
        const int mt = job / 6, nt = job % 6;
        v8f c = {};
#pragma unroll
        for (int k0 = 0; k0 < PTOK; k0 += 32)
            c = WMMA_BF16(load_frag_a(s_attnh + mt * 16 * PTOK + k0, PTOK),
                          load_frag_b(s_vT    + nt * 16 * PTOK + k0, PTOK), c);
        const int col = nt * 16 + lr;
        const float lb = lepe_b[col];
#pragma unroll
        for (int j = 0; j < 8; j++) {
            const int row = mt * 16 + j + hf * 8;
            const int py = row >> 3, px = row & 7;
            float acc = lb;
#pragma unroll
            for (int ky = 0; ky < 3; ky++) {
                const int yy = py + ky - 1;
                if ((unsigned)yy < 8u) {
#pragma unroll
                    for (int kx = 0; kx < 3; kx++) {
                        const int xx = px + kx - 1;
                        if ((unsigned)xx < 8u)
                            acc += s_lw[col * 9 + ky * 3 + kx] *
                                   (float)s_xw[(yy * 8 + xx) * CDIM + col];
                    }
                }
            }
            s_o[row * CDIM + col] = (bf16)(c[j] + acc);
        }
    }
    __syncthreads();

    // ---- out = o @ Wp^T' + bp  -> scatter back to [B,C,H,W]
    const int b = n >> 10, gy = (n >> 5) & 31, gx = n & 31;
    for (int job = wave; job < 24; job += 4) {
        const int mt = job / 6, nt = job % 6;
        v8f c = {};
#pragma unroll
        for (int k0 = 0; k0 < CDIM; k0 += 32)
            c = WMMA_BF16(load_frag_a(s_o + mt * 16 * CDIM + k0, CDIM),
                          load_frag_b(Wh + 3 * 9216 + nt * 16 * CDIM + k0, CDIM), c);
        const int col = nt * 16 + lr;
        const float bb = bp[col];
#pragma unroll
        for (int j = 0; j < 8; j++) {
            const int row = mt * 16 + j + hf * 8;
            const int py = row >> 3, px = row & 7;
            const int h = gy * 8 + py, w = gx * 8 + px;
            out[((size_t)(b * CDIM + col) * HW + h) * HW + w] = c[j] + bb;
        }
    }
}

// ---------------------------------------------------------------- launcher
extern "C" void kernel_launch(void* const* d_in, const int* in_sizes, int n_in,
                              void* d_out, int out_size, void* d_ws, size_t ws_size,
                              hipStream_t stream) {
    const float* x      = (const float*)d_in[0];
    const float* Wq     = (const float*)d_in[1];
    const float* bq     = (const float*)d_in[2];
    const float* Wk     = (const float*)d_in[3];
    const float* bk     = (const float*)d_in[4];
    const float* Wv     = (const float*)d_in[5];
    const float* bv     = (const float*)d_in[6];
    const float* Wp     = (const float*)d_in[7];
    const float* bp     = (const float*)d_in[8];
    const float* lepe_w = (const float*)d_in[9];
    const float* lepe_b = (const float*)d_in[10];
    float* out = (float*)d_out;

    // workspace layout (256B aligned)
    char* ws = (char*)d_ws;
    bf16* dsxh      = (bf16*)(ws);                      // 4096*96*2      = 786432
    bf16* Wh        = (bf16*)(ws + 786432);             // 4*96*96*2      = 73728
    int*  topk_idx  = (int*)(ws + 860160);              // 4096*3*4       = 49152
    float* topk_wgt = (float*)(ws + 909312);            // 4096*3*4       = 49152
    bf16* xwh       = (bf16*)(ws + 958464);             // 4096*64*96*2   = 50331648

    pool_kernel<<<(CDIM * NWIN) / 256, 256, 0, stream>>>(x, dsxh);
    partition_kernel<<<(4 * CDIM * HW * HW) / (256 * 4), 256, 0, stream>>>(x, xwh);
    wconv_kernel<<<(4 * CDIM * CDIM) / 256, 256, 0, stream>>>(Wq, Wk, Wv, Wp, Wh);
    sim_topk_kernel<<<NWIN / 16, 128, 0, stream>>>(dsxh, topk_idx, topk_wgt);
    attn_kernel<<<NWIN, 128, 0, stream>>>(xwh, Wh, topk_idx, topk_wgt,
                                          bq, bk, bv, bp, lepe_w, lepe_b, out);
    (void)in_sizes; (void)n_in; (void)out_size; (void)ws_size;
}
